// Legendre_test_21371757265197
// MI455X (gfx1250) — compile-verified, matrix operand order checked by reading, not run on hardware
//
#include <hip/hip_runtime.h>
#include <hip/hip_bf16.h>

#define N_TOTAL 65536
#define D_DIM   512
#define K_DIM   256
#define O_DIM   512
#define DEG     8
#define MTILES  4            // 16-row M tiles per workgroup (64 rows)
#define MROWS   (MTILES*16)

typedef _Float16 v16h __attribute__((ext_vector_type(16)));
typedef _Float16 v8h  __attribute__((ext_vector_type(8)));
typedef _Float16 v4h  __attribute__((ext_vector_type(4)));
typedef float    v8f  __attribute__((ext_vector_type(8)));
typedef float    v4f  __attribute__((ext_vector_type(4)));

// LDS row pitches (halves), 16B-aligned and bank-skewed
#define ZLD  520   // z tile:    64 rows x 512 cols f16
#define OLDH 264   // out8 tile: 64 rows x 256 cols f16

static __device__ __forceinline__ v16h ld_frag_h(const _Float16* p) {
  v8h lo = *(const v8h*)p;
  v8h hi = *(const v8h*)(p + 8);
  return __builtin_shufflevector(lo, hi, 0,1,2,3,4,5,6,7,8,9,10,11,12,13,14,15);
}

static __device__ __forceinline__ v8f v8f_splat(float x) {
  v8f r;
#pragma unroll
  for (int i = 0; i < 8; ++i) r[i] = x;
  return r;
}

// Prep: convert T [8,256,512] and C_w [512,256] from f32 to f16 in workspace
__global__ void legendre_cvt_kernel(const float* __restrict__ T,
                                    const float* __restrict__ Cw,
                                    _Float16* __restrict__ Tf,
                                    _Float16* __restrict__ Cwf) {
  const int idx = blockIdx.x * 256 + threadIdx.x;
  const int nT = DEG * K_DIM * D_DIM;
  if (idx < nT) {
    Tf[idx] = (_Float16)T[idx];
  } else {
    const int j = idx - nT;
    if (j < O_DIM * K_DIM) Cwf[j] = (_Float16)Cw[j];
  }
}

// Fused: 8 GEMMs (z @ T_n.T) -> Legendre recurrence -> out8 @ C_w.T + b
__global__ __launch_bounds__(256)
void legendre_main_kernel(const float*    __restrict__ z,
                          const float*    __restrict__ T0,
                          const _Float16* __restrict__ Tf,
                          const _Float16* __restrict__ Cwf,
                          const float*    __restrict__ Cb,
                          float*          __restrict__ out) {
  __shared__ __align__(16) _Float16 zsh[MROWS * ZLD];   // 66560 B
  __shared__ __align__(16) _Float16 osh[MROWS * OLDH];  // 33792 B

  const int t    = threadIdx.x;
  const int wave = t >> 5;
  const int lane = t & 31;
  const int lhi  = lane >> 4;   // 0/1: which 16-wide K half
  const int llo  = lane & 15;   // row (A) / col (B,C,D) within tile
  const int m0   = blockIdx.x * MROWS;

  // ---- Stage A: load z[64 x 512] f32, convert to f16, stage in LDS ----
  {
    const int row = t >> 2;            // 0..63
    const int c0  = (t & 3) * 128;     // 128 contiguous cols per thread
    const float* zp = z + (size_t)(m0 + row) * D_DIM + c0;
    _Float16* dp = zsh + row * ZLD + c0;
#pragma unroll
    for (int j = 0; j < 32; ++j) {
      v4f v = *(const v4f*)(zp + j * 4);
      v4h h;
      h.x = (_Float16)v.x; h.y = (_Float16)v.y;
      h.z = (_Float16)v.z; h.w = (_Float16)v.w;
      *(v4h*)(dp + j * 4) = h;
    }
  }
  __syncthreads();

  // per-wave invariant bases
  const _Float16* zbase = zsh + llo * ZLD + lhi * 16;

  // ---- Stage B: per wave, 2 K-blocks of 16; degrees processed in 2 passes
  //      of 4 so the recurrence state (p1,p2) replaces finished accumulators.
  for (int kbi = 0; kbi < 2; ++kbi) {
    const int kb = wave * 2 + kbi;     // K-block index (0..15)
    v8f p1[MTILES], p2[MTILES];

#pragma unroll
    for (int pass = 0; pass < 2; ++pass) {
      const _Float16* tbase = Tf + (size_t)(kb * 16 + llo) * D_DIM + lhi * 16
                                 + (size_t)(pass * 4) * (K_DIM * D_DIM);
      v8f acc[MTILES][4];
#pragma unroll
      for (int m = 0; m < MTILES; ++m)
#pragma unroll
        for (int d = 0; d < 4; ++d) acc[m][d] = v8f_splat(0.0f);

#pragma unroll 1
      for (int ch = 0; ch < 16; ++ch) {        // D chunks of 32 (runtime loop:
        v16h a[MTILES];                        //  bounds scheduler lookahead)
#pragma unroll
        for (int m = 0; m < MTILES; ++m)
          a[m] = ld_frag_h(zbase + m * (16 * ZLD) + ch * 32);
#pragma unroll
        for (int dd = 0; dd < 4; ++dd) {
          const v16h b = ld_frag_h(tbase + (size_t)dd * (K_DIM * D_DIM) + ch * 32);
#pragma unroll
          for (int m = 0; m < MTILES; ++m)
            acc[m][dd] = __builtin_amdgcn_wmma_f32_16x16x32_f16(
                false, a[m], false, b, (short)0, acc[m][dd], false, false);
        }
      }

      // advance the Legendre recurrence through this pass's degrees
      if (pass == 0) {
        const float t0v = T0[kb * 16 + llo];
#pragma unroll
        for (int m = 0; m < MTILES; ++m) {
          p2[m] = v8f_splat(t0v);    // degree 0 (broadcast)
          p1[m] = acc[m][0];         // degree 1
#pragma unroll
          for (int n = 2; n <= 4; ++n) {
            const float an = (2.0f * n - 1.0f) / (float)n;
            const float bn = ((float)n - 1.0f) / (float)n;
            v8f cur = (v8f_splat(an) * acc[m][n - 1]) * p1[m]
                      - v8f_splat(bn) * p2[m];
            p2[m] = p1[m];
            p1[m] = cur;
          }
        }
      } else {
#pragma unroll
        for (int m = 0; m < MTILES; ++m) {
#pragma unroll
          for (int n = 5; n <= 8; ++n) {
            const float an = (2.0f * n - 1.0f) / (float)n;
            const float bn = ((float)n - 1.0f) / (float)n;
            v8f cur = (v8f_splat(an) * acc[m][n - 5]) * p1[m]
                      - v8f_splat(bn) * p2[m];
            p2[m] = p1[m];
            p1[m] = cur;
          }
        }
      }
    }

    // store out8 tiles (f16) to LDS: C/D layout row = vgpr + 8*lhi, col = llo
#pragma unroll
    for (int m = 0; m < MTILES; ++m)
#pragma unroll
      for (int v = 0; v < 8; ++v)
        osh[(m * 16 + lhi * 8 + v) * OLDH + kb * 16 + llo] = (_Float16)p1[m][v];
  }
  __syncthreads();

  // ---- Stage C: out[64 x 512] += out8[64 x 256] @ C_w.T, wave owns 64 cols ----
  v8f oacc[MTILES][4];
#pragma unroll
  for (int m = 0; m < MTILES; ++m)
#pragma unroll
    for (int ot = 0; ot < 4; ++ot) oacc[m][ot] = v8f_splat(0.0f);

#pragma unroll 1
  for (int kc = 0; kc < 8; ++kc) {             // K chunks of 32 (runtime loop)
    const int c0 = kc * 32 + lhi * 16;
    v16h aO[MTILES];
#pragma unroll
    for (int m = 0; m < MTILES; ++m)
      aO[m] = ld_frag_h(osh + (m * 16 + llo) * OLDH + c0);
#pragma unroll
    for (int ot = 0; ot < 4; ++ot) {
      const _Float16* cp = Cwf + (size_t)(wave * 64 + ot * 16 + llo) * K_DIM + c0;
      const v16h b = ld_frag_h(cp);
#pragma unroll
      for (int m = 0; m < MTILES; ++m)
        oacc[m][ot] = __builtin_amdgcn_wmma_f32_16x16x32_f16(
            false, aO[m], false, b, (short)0, oacc[m][ot], false, false);
    }
  }

  // write f32 output with bias
#pragma unroll
  for (int m = 0; m < MTILES; ++m) {
#pragma unroll
    for (int ot = 0; ot < 4; ++ot) {
      const int col = wave * 64 + ot * 16 + llo;
      const float bias = Cb[col];
      float* op = out + (size_t)(m0 + m * 16 + lhi * 8) * O_DIM + col;
#pragma unroll
      for (int v = 0; v < 8; ++v)
        op[(size_t)v * O_DIM] = oacc[m][ot][v] + bias;
    }
  }
}

extern "C" void kernel_launch(void* const* d_in, const int* in_sizes, int n_in,
                              void* d_out, int out_size, void* d_ws, size_t ws_size,
                              hipStream_t stream) {
  const float* z  = (const float*)d_in[0];
  const float* T0 = (const float*)d_in[1];
  const float* T  = (const float*)d_in[2];
  const float* Cw = (const float*)d_in[3];
  const float* Cb = (const float*)d_in[4];
  float* out = (float*)d_out;

  _Float16* Tf  = (_Float16*)d_ws;                       // 8*256*512 halves
  _Float16* Cwf = Tf + (size_t)DEG * K_DIM * D_DIM;      // 512*256 halves

  const int ncvt = DEG * K_DIM * D_DIM + O_DIM * K_DIM;
  legendre_cvt_kernel<<<(ncvt + 255) / 256, 256, 0, stream>>>(T, Cw, Tf, Cwf);
  legendre_main_kernel<<<N_TOTAL / MROWS, 256, 0, stream>>>(z, T0, Tf, Cwf, Cb, out);
}
